// GlobalFluxTracker_79328045957538
// MI455X (gfx1250) — matrix operand order repaired and checked by reading, not compiled, not run on hardware
//
#include <hip/hip_runtime.h>
#include <math.h>

// ---------------------------------------------------------------------------
// GlobalFluxTracker on MI455X (gfx1250, wave32, WMMA, LDS-staged A panels)
//   B=8, T=4096, D=256, N=B*T=32768, 2D=512, 3D=768
// Pipeline:
//   K0 prep      : f32->bf16 weight conversion (W_in; W_out||W_gate), lam = -softplus+i*im
//   K1 gemm_in   : block = one 16-token M-tile; A panel (16x512) staged to LDS as bf16;
//                  each wave: 2 complex-pair N-tiles -> 4 WMMAs per A fragment; fused
//                  decay/forcing epilogue (fast TRANS ops) -> buf_decay, buf_X
//   K2 scan1     : per (b,chunk,d) chunk-local complex scan -> aggregates (A,Bagg)
//   K3 carry     : per (b,d) sequential combine of 16 chunk aggregates -> carries
//   K4 scan3     : replay chunk scan seeded with carry, write s_cat in place over buf_X
//   K5 gemm_out  : block = one M-tile; A panel staged to LDS; each wave: 6 N-tiles ->
//                  6 WMMAs per A fragment; fused bias + sigmoid-gate -> d_out [B,T,3D]
// ---------------------------------------------------------------------------

typedef __attribute__((ext_vector_type(16))) __bf16        v16bf;
typedef __attribute__((ext_vector_type(8)))  float         v8f;
typedef __attribute__((ext_vector_type(4)))  unsigned int  v4u;

#define GFT_B   8
#define GFT_T   4096
#define GFT_D   256
#define GFT_N   (GFT_B * GFT_T)      // 32768 tokens
#define GFT_2D  (2 * GFT_D)          // 512
#define GFT_3D  (3 * GFT_D)          // 768
#define GFT_TC  256                  // scan chunk length
#define GFT_NC  (GFT_T / GFT_TC)     // 16 chunks per (b,d)
#define GFT_EPS 1e-7f

// ---------------- K0: weight conversion + lambda precompute ----------------
__global__ __launch_bounds__(256)
void gft_prep(const float* __restrict__ Win, const float* __restrict__ Wout,
              const float* __restrict__ Wgate,
              const float* __restrict__ dre, const float* __restrict__ dim,
              __bf16* __restrict__ wbf_in, __bf16* __restrict__ wbf_og,
              float* __restrict__ lam_re, float* __restrict__ lam_im)
{
    const int stride = gridDim.x * blockDim.x;
    const int i0 = blockIdx.x * blockDim.x + threadIdx.x;
    for (int i = i0; i < GFT_2D * GFT_2D; i += stride)
        wbf_in[i] = (__bf16)Win[i];
    for (int i = i0; i < GFT_2D * GFT_2D; i += stride)
        wbf_og[i] = (__bf16)Wout[i];                       // rows 0..511
    for (int i = i0; i < GFT_D * GFT_2D; i += stride)
        wbf_og[GFT_2D * GFT_2D + i] = (__bf16)Wgate[i];    // rows 512..767
    if (i0 < GFT_D) {
        float x  = dre[i0];
        float sp = (x > 20.f) ? x : log1pf(expf(x));       // stable softplus (tiny, keep exact)
        lam_re[i0] = -sp;
        lam_im[i0] = dim[i0];
    }
}

// A-fragment from the LDS bf16 panel (16x32 A tile, contiguous K order matching B):
// lane L: M = L%16; K = 16*(L/16) + s for elems s = 0..15 (contiguous);
// v16bf elems 0..7 = first 8, 8..15 = next 8.  Two ds_load_b128.
__device__ inline v16bf gft_afrag_lds(const __bf16* __restrict__ panel,
                                      int l16, int k0, int half)
{
    union { v4u q[2]; v16bf v; } u;
    const __bf16* lp = panel + l16 * GFT_2D + k0 + 16 * half;
    u.q[0] = *reinterpret_cast<const v4u*>(lp);        // K 0..7 of half (8 bf16 = 16B)
    u.q[1] = *reinterpret_cast<const v4u*>(lp + 8);    // K 8..15 of half
    return u.v;
}

// ---------------- K1: input GEMM + fused decay/forcing epilogue ------------
__device__ inline void gft_forcing_epilogue(const v8f& acr, const v8f& aci,
                                            int m0, int half, int d,
                                            const float* __restrict__ dt,
                                            const float* __restrict__ lam_re,
                                            const float* __restrict__ lam_im,
                                            const float* __restrict__ b_in,
                                            float* __restrict__ buf_decay,
                                            float* __restrict__ buf_X)
{
    const float lre = lam_re[d];
    const float lim = lam_im[d];
    const float bir = b_in[d];
    const float bii = b_in[d + GFT_D];
#pragma unroll
    for (int r = 0; r < 8; ++r) {
        const int   m   = m0 + r + 8 * half;          // C layout: M = r + 8*(lane>=16)
        const float dtv = dt[m];                      // DT_REF = 1.0
        const float a   = lre * dtv;
        const float b   = lim * dtv;
        const float ea  = __expf(a);                  // v_exp_f32 (TRANS)
        const float cb  = __cosf(b);                  // v_cos_f32
        const float sb  = __sinf(b);                  // v_sin_f32
        const float dre = ea * cb, dimv = ea * sb;    // decay = exp(lam*dt)
        const float sh  = __sinf(0.5f * b);
        const float emr = (ea - 1.f) * cb - 2.f * sh * sh;   // Re cexpm1
        const float emi = ea * sb;                            // Im cexpm1
        const float sgn = ((a + GFT_EPS) >= 0.f) ? 1.f : -1.f;
        const float dnr = a + GFT_EPS * sgn;
        const float inv = 1.f / (dnr * dnr + b * b);
        const float fre = (emr * dnr + emi * b) * inv;        // cexpm1 / denom
        const float fim = (emi * dnr - emr * b) * inv;
        const float xr  = acr[r] + bir;
        const float xi  = aci[r] + bii;
        const size_t o  = (size_t)m * GFT_2D + d;
        buf_decay[o]         = dre;
        buf_decay[o + GFT_D] = dimv;
        buf_X[o]             = xr * fre - xi * fim;
        buf_X[o + GFT_D]     = xr * fim + xi * fre;
    }
}

// Block = one 16-token M-tile; wave w handles complex pairs {2w, 2w+1}.
__global__ __launch_bounds__(256)
void gft_gemm_in(const float* __restrict__ x_re, const float* __restrict__ x_im,
                 const float* __restrict__ dt,   const float* __restrict__ b_in,
                 const __bf16* __restrict__ wbf, const float* __restrict__ lam_re,
                 const float* __restrict__ lam_im,
                 float* __restrict__ buf_decay,  float* __restrict__ buf_X)
{
    __shared__ __bf16 sA[16 * GFT_2D];                // 16 KB A panel
    const int tid = threadIdx.x;
    const int m0  = blockIdx.x * 16;

    // Stage x_cat[m0..m0+15][0..511] -> bf16 LDS (float4 global loads, coalesced)
    for (int idx = tid; idx < 16 * (GFT_2D / 4); idx += 256) {
        const int rrow = idx >> 7;                    // row 0..15
        const int k    = (idx & 127) * 4;             // col 0..508 step 4
        const float4 v = (k < GFT_D)
            ? *reinterpret_cast<const float4*>(x_re + (size_t)(m0 + rrow) * GFT_D + k)
            : *reinterpret_cast<const float4*>(x_im + (size_t)(m0 + rrow) * GFT_D + (k - GFT_D));
        const int o = rrow * GFT_2D + k;
        sA[o + 0] = (__bf16)v.x; sA[o + 1] = (__bf16)v.y;
        sA[o + 2] = (__bf16)v.z; sA[o + 3] = (__bf16)v.w;
    }
    __syncthreads();

    const int lane = tid & 31, wid = tid >> 5;
    const int half = lane >> 4, l16 = lane & 15;
    const int d0a  = wid * 32;                        // first complex-pair base
    const int d0b  = d0a + 16;                        // second complex-pair base

    v8f ar0 = {}, ai0 = {}, ar1 = {}, ai1 = {};
    for (int k0 = 0; k0 < GFT_2D; k0 += 32) {
        const v16bf a = gft_afrag_lds(sA, l16, k0, half);

        const size_t kb = (size_t)k0 + 16 * half;
        const __bf16* br0 = wbf + (size_t)(d0a + l16)          * GFT_2D + kb;
        const __bf16* bi0 = wbf + (size_t)(d0a + GFT_D + l16)  * GFT_2D + kb;
        const __bf16* br1 = wbf + (size_t)(d0b + l16)          * GFT_2D + kb;
        const __bf16* bi1 = wbf + (size_t)(d0b + GFT_D + l16)  * GFT_2D + kb;
        __builtin_prefetch((const void*)(br0 + 32), 0, 1);     // next K step
        const v16bf f0 = *reinterpret_cast<const v16bf*>(br0);
        const v16bf f1 = *reinterpret_cast<const v16bf*>(bi0);
        const v16bf f2 = *reinterpret_cast<const v16bf*>(br1);
        const v16bf f3 = *reinterpret_cast<const v16bf*>(bi1);

        ar0 = __builtin_amdgcn_wmma_f32_16x16x32_bf16(false, a, false, f0, (short)0, ar0, false, false);
        ai0 = __builtin_amdgcn_wmma_f32_16x16x32_bf16(false, a, false, f1, (short)0, ai0, false, false);
        ar1 = __builtin_amdgcn_wmma_f32_16x16x32_bf16(false, a, false, f2, (short)0, ar1, false, false);
        ai1 = __builtin_amdgcn_wmma_f32_16x16x32_bf16(false, a, false, f3, (short)0, ai1, false, false);
    }

    gft_forcing_epilogue(ar0, ai0, m0, half, d0a + l16, dt, lam_re, lam_im, b_in, buf_decay, buf_X);
    gft_forcing_epilogue(ar1, ai1, m0, half, d0b + l16, dt, lam_re, lam_im, b_in, buf_decay, buf_X);
}

// ---------------- K2: chunk-local scan aggregates --------------------------
// agg planes: [0]=A.re [1]=A.im [2]=Bagg.re [3]=Bagg.im, each B*NC*D entries
__global__ __launch_bounds__(256)
void gft_scan1(const float* __restrict__ buf_decay, const float* __restrict__ buf_X,
               float* __restrict__ agg)
{
    const int tid   = blockIdx.x * 256 + threadIdx.x;   // 0..32767
    const int d     = tid & (GFT_D - 1);
    const int chunk = (tid >> 8) & (GFT_NC - 1);
    const int b     = tid >> 12;
    size_t base = ((size_t)b * GFT_T + (size_t)chunk * GFT_TC) * GFT_2D + d;
    float Are = 1.f, Aim = 0.f, sre = 0.f, sim = 0.f;
    for (int t = 0; t < GFT_TC; ++t) {
        const float are = buf_decay[base], aim = buf_decay[base + GFT_D];
        const float xre = buf_X[base],     xim = buf_X[base + GFT_D];
        const float nsr = are * sre - aim * sim + xre;
        const float nsi = are * sim + aim * sre + xim;
        sre = nsr; sim = nsi;
        const float nAr = are * Are - aim * Aim;
        const float nAi = are * Aim + aim * Are;
        Are = nAr; Aim = nAi;
        base += GFT_2D;
    }
    const int P = GFT_B * GFT_NC * GFT_D;               // 32768
    agg[tid] = Are; agg[P + tid] = Aim; agg[2 * P + tid] = sre; agg[3 * P + tid] = sim;
}

// ---------------- K3: carry propagation across chunks ----------------------
__global__ __launch_bounds__(256)
void gft_carry(const float* __restrict__ agg, float* __restrict__ carry)
{
    const int tid = blockIdx.x * 256 + threadIdx.x;     // 0..2047 = b*D + d
    const int d = tid & (GFT_D - 1);
    const int b = tid >> 8;
    const int P = GFT_B * GFT_NC * GFT_D;
    float cre = 0.f, cim = 0.f;
#pragma unroll
    for (int c = 0; c < GFT_NC; ++c) {
        const int o = (b * GFT_NC + c) * GFT_D + d;
        carry[o] = cre; carry[P + o] = cim;             // carry INTO chunk c
        const float Are = agg[o],         Aim = agg[P + o];
        const float Bre = agg[2 * P + o], Bim = agg[3 * P + o];
        const float nre = Are * cre - Aim * cim + Bre;
        const float nim = Are * cim + Aim * cre + Bim;
        cre = nre; cim = nim;
    }
}

// ---------------- K4: seeded replay, write s_cat in place over buf_X -------
__global__ __launch_bounds__(256)
void gft_scan3(const float* __restrict__ buf_decay, float* __restrict__ buf_X,
               const float* __restrict__ carry)
{
    const int tid   = blockIdx.x * 256 + threadIdx.x;
    const int d     = tid & (GFT_D - 1);
    const int chunk = (tid >> 8) & (GFT_NC - 1);
    const int b     = tid >> 12;
    const int P     = GFT_B * GFT_NC * GFT_D;
    const int co    = (b * GFT_NC + chunk) * GFT_D + d;
    float sre = carry[co], sim = carry[P + co];
    size_t base = ((size_t)b * GFT_T + (size_t)chunk * GFT_TC) * GFT_2D + d;
    for (int t = 0; t < GFT_TC; ++t) {
        const float are = buf_decay[base], aim = buf_decay[base + GFT_D];
        const float xre = buf_X[base],     xim = buf_X[base + GFT_D];
        const float nsr = are * sre - aim * sim + xre;
        const float nsi = are * sim + aim * sre + xim;
        sre = nsr; sim = nsi;
        buf_X[base]         = sre;                      // s_cat re plane
        buf_X[base + GFT_D] = sim;                      // s_cat im plane
        base += GFT_2D;
    }
}

// ---------------- K5: output GEMM (W_out || W_gate), fused gate ------------
// Block = one M-tile; wave w handles N-tiles {6w .. 6w+5} of the 48 (768 cols).
__global__ __launch_bounds__(256)
void gft_gemm_out(const float* __restrict__ s_cat, const __bf16* __restrict__ wbf_og,
                  const float* __restrict__ b_out, const float* __restrict__ b_gate,
                  float* __restrict__ out)
{
    __shared__ __bf16 sA[16 * GFT_2D];                  // 16 KB
    const int tid = threadIdx.x;
    const int m0  = blockIdx.x * 16;

    for (int idx = tid; idx < 16 * (GFT_2D / 4); idx += 256) {
        const int rrow = idx >> 7;
        const int k    = (idx & 127) * 4;
        const float4 v = *reinterpret_cast<const float4*>(
            s_cat + (size_t)(m0 + rrow) * GFT_2D + k);
        const int o = rrow * GFT_2D + k;
        sA[o + 0] = (__bf16)v.x; sA[o + 1] = (__bf16)v.y;
        sA[o + 2] = (__bf16)v.z; sA[o + 3] = (__bf16)v.w;
    }
    __syncthreads();

    const int lane = tid & 31, wid = tid >> 5;
    const int half = lane >> 4, l16 = lane & 15;

    v8f acc[6] = {};
    for (int k0 = 0; k0 < GFT_2D; k0 += 32) {
        const v16bf a = gft_afrag_lds(sA, l16, k0, half);
        const size_t kb = (size_t)k0 + 16 * half;
        __builtin_prefetch((const void*)(wbf_og + (size_t)(wid * 96 + l16) * GFT_2D + kb + 32), 0, 1);
#pragma unroll
        for (int i = 0; i < 6; ++i) {
            const int n0 = (wid * 6 + i) * 16;
            const v16bf bf = *reinterpret_cast<const v16bf*>(
                wbf_og + (size_t)(n0 + l16) * GFT_2D + kb);
            acc[i] = __builtin_amdgcn_wmma_f32_16x16x32_bf16(false, a, false, bf,
                                                             (short)0, acc[i], false, false);
        }
    }

#pragma unroll
    for (int i = 0; i < 6; ++i) {
        const int   j    = (wid * 6 + i) * 16 + l16;    // 0..767 output column
        const float bias = (j < GFT_2D) ? b_out[j] : b_gate[j - GFT_2D];
#pragma unroll
        for (int r = 0; r < 8; ++r) {
            const int m = m0 + r + 8 * half;
            float v = acc[i][r] + bias;
            if (j >= GFT_2D)                            // gate columns
                v = 0.98f / (1.f + __expf(-v)) + 0.01f;
            out[(size_t)m * GFT_3D + j] = v;
        }
    }
}

// ---------------------------------------------------------------------------
extern "C" void kernel_launch(void* const* d_in, const int* in_sizes, int n_in,
                              void* d_out, int out_size, void* d_ws, size_t ws_size,
                              hipStream_t stream)
{
    (void)in_sizes; (void)n_in; (void)out_size; (void)ws_size;
    const float* x_re   = (const float*)d_in[0];
    const float* x_im   = (const float*)d_in[1];
    const float* dt     = (const float*)d_in[2];
    const float* dec_re = (const float*)d_in[3];
    const float* dec_im = (const float*)d_in[4];
    const float* W_in   = (const float*)d_in[5];
    const float* b_in   = (const float*)d_in[6];
    const float* W_out  = (const float*)d_in[7];
    const float* b_out  = (const float*)d_in[8];
    const float* W_gate = (const float*)d_in[9];
    const float* b_gate = (const float*)d_in[10];
    float* out = (float*)d_out;

    // Workspace carve-up (256B aligned chunks)
    char* ws = (char*)d_ws;
    size_t off = 0;
    auto carve = [&](size_t bytes) {
        char* p = ws + off;
        off += (bytes + 255) & ~(size_t)255;
        return p;
    };
    __bf16* wbf_in    = (__bf16*)carve((size_t)GFT_2D * GFT_2D * sizeof(__bf16)); // 512 KB
    __bf16* wbf_og    = (__bf16*)carve((size_t)GFT_3D * GFT_2D * sizeof(__bf16)); // 768 KB
    float*  lam_re    = (float*) carve(GFT_D * sizeof(float));
    float*  lam_im    = (float*) carve(GFT_D * sizeof(float));
    float*  buf_decay = (float*) carve((size_t)GFT_N * GFT_2D * sizeof(float));   // 64 MB
    float*  buf_X     = (float*) carve((size_t)GFT_N * GFT_2D * sizeof(float));   // 64 MB
    float*  agg       = (float*) carve((size_t)4 * GFT_B * GFT_NC * GFT_D * sizeof(float));
    float*  carry     = (float*) carve((size_t)2 * GFT_B * GFT_NC * GFT_D * sizeof(float));

    // K0: weights -> bf16, lambda
    gft_prep<<<1024, 256, 0, stream>>>(W_in, W_out, W_gate, dec_re, dec_im,
                                       wbf_in, wbf_og, lam_re, lam_im);

    // K1: input GEMM + fused forcing: one block per 16-token M-tile
    gft_gemm_in<<<GFT_N / 16, 256, 0, stream>>>(x_re, x_im, dt, b_in, wbf_in,
                                                lam_re, lam_im, buf_decay, buf_X);

    // K2: chunk aggregates: B*NC*D = 32768 threads
    gft_scan1<<<GFT_B * GFT_NC * GFT_D / 256, 256, 0, stream>>>(buf_decay, buf_X, agg);

    // K3: carries: B*D = 2048 threads
    gft_carry<<<GFT_B * GFT_D / 256, 256, 0, stream>>>(agg, carry);

    // K4: seeded replay -> s_cat in place
    gft_scan3<<<GFT_B * GFT_NC * GFT_D / 256, 256, 0, stream>>>(buf_decay, buf_X, carry);

    // K5: output GEMM: one block per M-tile, 48 N-tiles across 8 waves
    gft_gemm_out<<<GFT_N / 16, 256, 0, stream>>>(buf_X, wbf_og, b_out, b_gate, out);
}
